// PyramidMamba_82282983456964
// MI455X (gfx1250) — matrix operand order repaired and checked by reading, not compiled, not run on hardware
//
#include <hip/hip_runtime.h>

// ---------------- constants ----------------
#define B_   4
#define CIN_ 512
#define H_   16
#define W_   16
#define L_   256
#define DIM_ 128
#define DM_  1024
#define DI_  2048
#define K4_  4
#define NS_  16
#define R_   64
#define E2_  4096

typedef __bf16 bf16_t;
typedef __attribute__((ext_vector_type(8)))  __bf16 v8bf;
typedef __attribute__((ext_vector_type(16))) __bf16 v16bf;
typedef __attribute__((ext_vector_type(8)))  float  v8f;

__device__ __forceinline__ bf16_t to_bf16(float f) {
    union { float f; unsigned u; } in; in.f = f;
    unsigned r = in.u + 0x7FFFu + ((in.u >> 16) & 1u);   // round-to-nearest-even
    union { unsigned short s; bf16_t b; } out;
    out.s = (unsigned short)(r >> 16);
    return out.b;
}
__device__ __forceinline__ float bf2f(bf16_t b) {
    union { bf16_t b; unsigned short s; } u; u.b = b;
    return __uint_as_float(((unsigned)u.s) << 16);
}

// gfx1250 async global->LDS copy (ASYNCcnt-tracked). lds_off = workgroup LDS byte offset.
__device__ __forceinline__ void async_g2l_b128(unsigned lds_off, const void* gaddr) {
    asm volatile("global_load_async_to_lds_b128 %0, %1, off"
                 :: "v"(lds_off), "v"(gaddr) : "memory");
}
__device__ __forceinline__ void wait_async_le3() {
#if __has_builtin(__builtin_amdgcn_s_wait_asynccnt)
    __builtin_amdgcn_s_wait_asynccnt(3);
#else
    asm volatile("s_wait_asynccnt 3" ::: "memory");
#endif
}
__device__ __forceinline__ void wait_async_le0() {
#if __has_builtin(__builtin_amdgcn_s_wait_asynccnt)
    __builtin_amdgcn_s_wait_asynccnt(0);
#else
    asm volatile("s_wait_asynccnt 0" ::: "memory");
#endif
}

// ---------------- generic f32 -> bf16 convert ----------------
__global__ void pm_cvt_bf16(const float* __restrict__ in, bf16_t* __restrict__ out, int n) {
    int t = blockIdx.x * blockDim.x + threadIdx.x;
    if (t < n) out[t] = to_bf16(in[t]);
}

// =====================================================================
// WMMA bf16 GEMM v1 (direct-from-global, one 16x16 tile per wave).
// Used only for N=96 (x_proj). C[m][n] = sum_k A[m*lda+k]*B[n*ldb+k].
// =====================================================================
__global__ void __launch_bounds__(256)
pm_gemm_bf16(const bf16_t* __restrict__ A, const bf16_t* __restrict__ Bm,
             float* __restrict__ Cm, int M, int N, int Kd, int lda, int ldb) {
    int tiles_n = N >> 4;
    int total   = (M >> 4) * tiles_n;
    int tile    = blockIdx.x * blockDim.y + threadIdx.y;
    if (tile >= total) return;                 // wave-uniform exit
    int tm = tile / tiles_n, tn = tile % tiles_n;
    int lane = threadIdx.x;
    int lmod = lane & 15;
    bool hi  = lane >= 16;

    const bf16_t* arow = A  + (size_t)(tm * 16 + lmod) * lda + (hi ? 8 : 0);
    const bf16_t* brow = Bm + (size_t)(tn * 16 + lmod) * ldb + (hi ? 8 : 0);

    v8f acc = {};
    for (int k0 = 0; k0 < Kd; k0 += 32) {
        if (k0 + 64 <= Kd) {
            __builtin_prefetch(arow + k0 + 32, 0, 1);
            __builtin_prefetch(brow + k0 + 32, 0, 1);
        }
        v8bf a_lo = *(const v8bf*)(arow + k0);
        v8bf a_hi = *(const v8bf*)(arow + k0 + 16);
        v8bf b_lo = *(const v8bf*)(brow + k0);
        v8bf b_hi = *(const v8bf*)(brow + k0 + 16);
        v16bf a, b;
#pragma unroll
        for (int i = 0; i < 8; i++) {
            a[i] = a_lo[i]; a[i + 8] = a_hi[i];
            b[i] = b_lo[i]; b[i + 8] = b_hi[i];
        }
        acc = __builtin_amdgcn_wmma_f32_16x16x32_bf16(false, a, false, b,
                                                      (short)0, acc, false, false);
    }
    int ncol  = tn * 16 + lmod;
    int mbase = tm * 16 + (hi ? 8 : 0);
#pragma unroll
    for (int v = 0; v < 8; v++)
        Cm[(size_t)(mbase + v) * N + ncol] = acc[v];
}

// =====================================================================
// WMMA bf16 GEMM v2: LDS-staged via async copies, double buffered.
// Block = 256 threads (8 wave32s) -> 128(M) x 64(N) macro-tile.
// Wave (wm 0..3, wn 0..1) computes a 32x32 region = 2x2 WMMA tiles.
// Per K-step(32): stage A 128x32 + B 64x32 (3 async b128 per thread),
// then 8 ds_load_b128 fragment reads feed 4 v_wmma per wave.
// LDS rows padded to 40 elements (80B) -> conflict-free 16-lane reads.
// Requires M%128==0, N%64==0, K%32==0.
// =====================================================================
#define PITCH_ 40

__global__ void __launch_bounds__(256)
pm_gemm_bf16_lds(const bf16_t* __restrict__ A, const bf16_t* __restrict__ Bm,
                 float* __restrict__ Cm, int M, int N, int Kd, int lda, int ldb) {
    __shared__ bf16_t As[2][128 * PITCH_];
    __shared__ bf16_t Bs[2][64 * PITCH_];

    int tiles_n = N >> 6;
    int bm = blockIdx.x / tiles_n;
    int bn = blockIdx.x % tiles_n;
    int lane = threadIdx.x;
    int wv   = threadIdx.y;
    int tid  = wv * 32 + lane;
    int wm = wv & 3, wn = wv >> 2;

    const size_t abase = (size_t)bm * 128;
    const size_t bbase = (size_t)bn * 64;

    auto stage = [&](int buf, int k0) {
        // A tile: 128 rows x 32 bf16 = 512 x 16B chunks, 2 per thread
#pragma unroll
        for (int c = 0; c < 2; c++) {
            int ch  = tid + c * 256;
            int row = ch >> 2, q = ch & 3;
            const void* g = A + (abase + row) * (size_t)lda + k0 + q * 8;
            unsigned loff = (unsigned)(size_t)&As[buf][row * PITCH_ + q * 8];
            async_g2l_b128(loff, g);
        }
        // B tile: 64 rows x 32 bf16 = 256 x 16B chunks, 1 per thread
        {
            int row = tid >> 2, q = tid & 3;
            const void* g = Bm + (bbase + row) * (size_t)ldb + k0 + q * 8;
            unsigned loff = (unsigned)(size_t)&Bs[buf][row * PITCH_ + q * 8];
            async_g2l_b128(loff, g);
        }
    };

    int lmod = lane & 15;
    bool hi  = lane >= 16;

    v8f acc00 = {}, acc01 = {}, acc10 = {}, acc11 = {};

    int nk = Kd >> 5;
    stage(0, 0);
    for (int ks = 0; ks < nk; ks++) {
        int cur = ks & 1;
        if (ks + 1 < nk) {
            __syncthreads();                 // everyone done reading buf cur^1
            stage(cur ^ 1, (ks + 1) << 5);   // prefetch next stage (3 async ops)
            wait_async_le3();                // current stage's 3 ops complete
        } else {
            wait_async_le0();
        }
        __syncthreads();                     // all lanes' staged data visible

        const bf16_t* abuf = &As[cur][(wm * 32) * PITCH_];
        const bf16_t* bbuf = &Bs[cur][(wn * 32) * PITCH_];

        v16bf af[2], bfr[2];
#pragma unroll
        for (int i = 0; i < 2; i++) {
            const bf16_t* ar = abuf + (i * 16 + lmod) * PITCH_ + (hi ? 8 : 0);
            v8bf alo = *(const v8bf*)(ar);
            v8bf ahi = *(const v8bf*)(ar + 16);
            const bf16_t* br = bbuf + (i * 16 + lmod) * PITCH_ + (hi ? 8 : 0);
            v8bf blo = *(const v8bf*)(br);
            v8bf bhi = *(const v8bf*)(br + 16);
#pragma unroll
            for (int j = 0; j < 8; j++) {
                af[i][j] = alo[j];  af[i][j + 8] = ahi[j];
                bfr[i][j] = blo[j]; bfr[i][j + 8] = bhi[j];
            }
        }
        acc00 = __builtin_amdgcn_wmma_f32_16x16x32_bf16(false, af[0], false, bfr[0], (short)0, acc00, false, false);
        acc01 = __builtin_amdgcn_wmma_f32_16x16x32_bf16(false, af[0], false, bfr[1], (short)0, acc01, false, false);
        acc10 = __builtin_amdgcn_wmma_f32_16x16x32_bf16(false, af[1], false, bfr[0], (short)0, acc10, false, false);
        acc11 = __builtin_amdgcn_wmma_f32_16x16x32_bf16(false, af[1], false, bfr[1], (short)0, acc11, false, false);
    }

    int mrow = bm * 128 + wm * 32 + (hi ? 8 : 0);
    int ncol = bn * 64  + wn * 32 + lmod;
#pragma unroll
    for (int v = 0; v < 8; v++) {
        Cm[(size_t)(mrow +  0 + v) * N + ncol +  0] = acc00[v];
        Cm[(size_t)(mrow +  0 + v) * N + ncol + 16] = acc01[v];
        Cm[(size_t)(mrow + 16 + v) * N + ncol +  0] = acc10[v];
        Cm[(size_t)(mrow + 16 + v) * N + ncol + 16] = acc11[v];
    }
}

// ---------------- pool pyramid ----------------
__global__ void pm_pool_conv(const float* __restrict__ p, const float* __restrict__ pw,
                             const float* __restrict__ pb, const float* __restrict__ g,
                             const float* __restrict__ bt, const float* __restrict__ m,
                             const float* __restrict__ v, float* __restrict__ q,
                             int ss, int li) {
    int blk = blockIdx.x;
    int b = blk / ss, ij = blk % ss;
    int d = threadIdx.x;
    const float* wrow = pw + ((size_t)(li * DIM_ + d)) * CIN_;
    const float* pp   = p + ((size_t)b * CIN_) * ss + ij;
    float acc = pb[li * DIM_ + d];
    for (int c = 0; c < CIN_; c++) acc += pp[(size_t)c * ss] * wrow[c];
    int gi = li * DIM_ + d;
    float inv = g[gi] * rsqrtf(v[gi] + 1e-5f);
    acc = acc * inv + (bt[gi] - m[gi] * inv);
    q[((size_t)b * ss + ij) * DIM_ + d] = fmaxf(acc, 0.f);
}

__global__ void pm_pool0_mean(const float* __restrict__ q0, float* __restrict__ y0) {
    int b = blockIdx.x, d = threadIdx.x;
    float s = 0.f;
    for (int l = 0; l < L_; l++) s += q0[((size_t)b * L_ + l) * DIM_ + d];
    y0[b * DIM_ + d] = s * (1.f / (float)L_);
}

__global__ void pm_adap_pool(const float* __restrict__ x, float* __restrict__ p, int s) {
    int bc = blockIdx.x;
    int b = bc / CIN_, c = bc % CIN_;
    int ij = threadIdx.x;
    int i = ij / s, j = ij % s;
    int hs = (i * H_) / s,  he = ((i + 1) * H_ + s - 1) / s;
    int ws = (j * W_) / s,  we = ((j + 1) * W_ + s - 1) / s;
    const float* xp = x + ((size_t)b * CIN_ + c) * L_;
    float acc = 0.f;
    for (int h = hs; h < he; h++)
        for (int w = ws; w < we; w++) acc += xp[h * W_ + w];
    p[((size_t)bc) * (s * s) + ij] = acc / (float)((he - hs) * (we - ws));
}

__global__ void pm_feat_assemble(const float* __restrict__ x, const float* __restrict__ y0,
                                 const float* __restrict__ q5, const float* __restrict__ q9,
                                 const float* __restrict__ q13, bf16_t* __restrict__ featb) {
    int pix = blockIdx.x;                     // b*256 + l
    int b = pix >> 8, l = pix & 255;
    int h = l >> 4, w = l & 15;
    int c = threadIdx.x;                      // 0..1023
    float val;
    if (c < CIN_) {
        val = x[((size_t)b * CIN_ + c) * L_ + l];
    } else if (c < CIN_ + DIM_) {
        val = y0[b * DIM_ + (c - CIN_)];
    } else {
        int li = (c - CIN_ - DIM_) >> 7;
        int d  = (c - CIN_ - DIM_) & 127;
        int s  = (li == 0) ? 5 : ((li == 1) ? 9 : 13);
        const float* q = (li == 0) ? q5 : ((li == 1) ? q9 : q13);
        float scale = (float)s / 16.f;
        float sh = fmaxf((h + 0.5f) * scale - 0.5f, 0.f);
        int h0 = (int)sh; if (h0 > s - 1) h0 = s - 1;
        float wh = sh - (float)h0;
        int h1 = h0 + 1; if (h1 > s - 1) h1 = s - 1;
        float sw = fmaxf((w + 0.5f) * scale - 0.5f, 0.f);
        int w0 = (int)sw; if (w0 > s - 1) w0 = s - 1;
        float ww = sw - (float)w0;
        int w1 = w0 + 1; if (w1 > s - 1) w1 = s - 1;
        int ss = s * s;
        const float* qb = q + ((size_t)b * ss) * DIM_ + d;
        float v00 = qb[(size_t)(h0 * s + w0) * DIM_];
        float v01 = qb[(size_t)(h0 * s + w1) * DIM_];
        float v10 = qb[(size_t)(h1 * s + w0) * DIM_];
        float v11 = qb[(size_t)(h1 * s + w1) * DIM_];
        val = (1.f - wh) * ((1.f - ww) * v00 + ww * v01)
            +        wh * ((1.f - ww) * v10 + ww * v11);
    }
    featb[(size_t)pix * DM_ + c] = to_bf16(val);
}

// ---------------- depthwise 3x3 conv + SiLU ----------------
__global__ void pm_dwconv_silu(const float* __restrict__ xz, const float* __restrict__ cw,
                               const float* __restrict__ cb, float* __restrict__ xc) {
    int t = blockIdx.x * blockDim.x + threadIdx.x;      // B*L*DI
    if (t >= B_ * L_ * DI_) return;
    int d = t & (DI_ - 1);
    int l = (t >> 11) & 255;
    int b = t >> 19;
    int h = l >> 4, w = l & 15;
    float acc = cb[d];
    const float* wv = cw + (size_t)d * 9;
#pragma unroll
    for (int kh = 0; kh < 3; kh++) {
        int hh = h + kh - 1;
        if (hh < 0 || hh > 15) continue;
#pragma unroll
        for (int kw = 0; kw < 3; kw++) {
            int wi = w + kw - 1;
            if (wi < 0 || wi > 15) continue;
            acc += xz[((size_t)(b * L_ + hh * W_ + wi)) * E2_ + d] * wv[kh * 3 + kw];
        }
    }
    xc[((size_t)(b * L_ + l)) * DI_ + d] = acc / (1.f + __expf(-acc));   // SiLU
}

// ---------------- build xs4 (k,b,l,d) bf16 ----------------
__global__ void pm_build_xs4(const float* __restrict__ xc, bf16_t* __restrict__ xs4) {
    int t = blockIdx.x * blockDim.x + threadIdx.x;      // K4*B*L*DI
    if (t >= K4_ * B_ * L_ * DI_) return;
    int d = t & (DI_ - 1);
    int l = (t >> 11) & 255;
    int b = (t >> 19) & 3;
    int k = t >> 21;
    int ls;
    if (k == 0) ls = l;
    else if (k == 1) { int wq = l >> 4, hq = l & 15; ls = hq * W_ + wq; }
    else if (k == 2) ls = (L_ - 1) - l;
    else { int lt = (L_ - 1) - l; int wq = lt >> 4, hq = lt & 15; ls = hq * W_ + wq; }
    xs4[t] = to_bf16(xc[((size_t)(b * L_ + ls)) * DI_ + d]);
}

// ---------------- delta = softplus(pre + dt_b) ----------------
__global__ void pm_softplus_bias(float* __restrict__ delta, const float* __restrict__ dtb) {
    int t = blockIdx.x * blockDim.x + threadIdx.x;
    if (t >= K4_ * B_ * L_ * DI_) return;
    int d = t & (DI_ - 1);
    int k = t >> 21;
    float xv = delta[t] + dtb[k * DI_ + d];
    delta[t] = (xv > 20.f) ? xv : log1pf(__expf(xv));
}

// ---------------- selective scan ----------------
__global__ void pm_scan(const bf16_t* __restrict__ xs4, const float* __restrict__ delta,
                        const float* __restrict__ xdbl, const float* __restrict__ A_logs,
                        float* __restrict__ outy) {
    int kb = blockIdx.y;
    int k = kb >> 2;
    int d = blockIdx.x * blockDim.x + threadIdx.x;
    float Ar[NS_], h[NS_];
    const float* al = A_logs + ((size_t)(k * DI_ + d)) * NS_;
#pragma unroll
    for (int n = 0; n < NS_; n++) { Ar[n] = -__expf(al[n]); h[n] = 0.f; }
    const bf16_t* up = xs4   + ((size_t)kb * L_) * DI_ + d;
    const float*  dp = delta + ((size_t)kb * L_) * DI_ + d;
    const float*  bc = xdbl  + ((size_t)kb * L_) * (R_ + 2 * NS_);
    float* yo = outy + (((size_t)kb) * DI_ + d) * L_;
    for (int l = 0; l < L_; l++) {
        float u  = bf2f(up[(size_t)l * DI_]);
        float dl = dp[(size_t)l * DI_];
        const float* bv = bc + (size_t)l * (R_ + 2 * NS_) + R_;
        float du = dl * u;
        float y = 0.f;
#pragma unroll
        for (int n = 0; n < NS_; n++) {
            h[n] = h[n] * __expf(dl * Ar[n]) + du * bv[n];
            y += h[n] * bv[NS_ + n];
        }
        yo[l] = y;
    }
}

// ---------------- combine + LN + SiLU(z) gate ----------------
__global__ void pm_combine_ln_gate(const float* __restrict__ outy, const float* __restrict__ xc,
                                   const float* __restrict__ xz, const float* __restrict__ Ds,
                                   const float* __restrict__ lng, const float* __restrict__ lnb,
                                   bf16_t* __restrict__ yln) {
    __shared__ float sh[DI_];
    __shared__ float red[512];
    int pix = blockIdx.x;
    int b = pix >> 8, l = pix & 255;
    int h = l >> 4, w = l & 15;
    int lwh = w * H_ + h;
    int tid = threadIdx.x;
    float lsum = 0.f, lsq = 0.f;
#pragma unroll
    for (int i = 0; i < 8; i++) {
        int d = i * 256 + tid;
        float y = outy[(((size_t)(0 * B_ + b)) * DI_ + d) * L_ + l]
                + outy[(((size_t)(1 * B_ + b)) * DI_ + d) * L_ + lwh]
                + outy[(((size_t)(2 * B_ + b)) * DI_ + d) * L_ + (L_ - 1 - l)]
                + outy[(((size_t)(3 * B_ + b)) * DI_ + d) * L_ + (L_ - 1 - lwh)];
        float dsum = Ds[d] + Ds[DI_ + d] + Ds[2 * DI_ + d] + Ds[3 * DI_ + d];
        y += dsum * xc[((size_t)pix) * DI_ + d];
        sh[d] = y; lsum += y; lsq += y * y;
    }
    red[tid] = lsum; red[256 + tid] = lsq;
    __syncthreads();
    for (int s = 128; s > 0; s >>= 1) {
        if (tid < s) { red[tid] += red[tid + s]; red[256 + tid] += red[256 + tid + s]; }
        __syncthreads();
    }
    float mean = red[0] * (1.f / (float)DI_);
    float var  = red[256] * (1.f / (float)DI_) - mean * mean;
    float inv  = rsqrtf(var + 1e-5f);
#pragma unroll
    for (int i = 0; i < 8; i++) {
        int d = i * 256 + tid;
        float y = (sh[d] - mean) * inv * lng[d] + lnb[d];
        float zv = xz[((size_t)pix) * E2_ + DI_ + d];
        y *= zv / (1.f + __expf(-zv));
        yln[(size_t)pix * DI_ + d] = to_bf16(y);
    }
}

// ---------------- cbr finish ----------------
__global__ void pm_cbr_finish(const float* __restrict__ pre, const float* __restrict__ cb,
                              const float* __restrict__ g, const float* __restrict__ bt,
                              const float* __restrict__ m, const float* __restrict__ v,
                              float* __restrict__ out) {
    int t = blockIdx.x * blockDim.x + threadIdx.x;
    if (t >= B_ * DIM_ * L_) return;
    int l = t & 255;
    int d = (t >> 8) & 127;
    int b = t >> 15;
    float xv = pre[((size_t)(b * L_ + l)) * DIM_ + d] + cb[d];
    float inv = g[d] * rsqrtf(v[d] + 1e-5f);
    xv = xv * inv + (bt[d] - m[d] * inv);
    out[t] = 0.5f * xv * (1.f + erff(xv * 0.70710678118654752f));
}

// ---------------- host ----------------
extern "C" void kernel_launch(void* const* d_in, const int* in_sizes, int n_in,
                              void* d_out, int out_size, void* d_ws, size_t ws_size,
                              hipStream_t stream) {
    const float* x        = (const float*)d_in[0];
    const float* pool_w   = (const float*)d_in[1];
    const float* pool_b   = (const float*)d_in[2];
    const float* bn_g     = (const float*)d_in[3];
    const float* bn_b     = (const float*)d_in[4];
    const float* bn_m     = (const float*)d_in[5];
    const float* bn_v     = (const float*)d_in[6];
    const float* in_proj  = (const float*)d_in[7];
    const float* conv_w   = (const float*)d_in[8];
    const float* conv_b   = (const float*)d_in[9];
    const float* x_proj   = (const float*)d_in[10];
    const float* dt_w     = (const float*)d_in[11];
    const float* dt_b     = (const float*)d_in[12];
    const float* A_logs   = (const float*)d_in[13];
    const float* Ds       = (const float*)d_in[14];
    const float* ln_g     = (const float*)d_in[15];
    const float* ln_b     = (const float*)d_in[16];
    const float* out_proj = (const float*)d_in[17];
    const float* cbr_w    = (const float*)d_in[18];
    const float* cbr_b    = (const float*)d_in[19];
    const float* cbr_g    = (const float*)d_in[20];
    const float* cbr_bt   = (const float*)d_in[21];
    const float* cbr_m    = (const float*)d_in[22];
    const float* cbr_v    = (const float*)d_in[23];

    char* base = (char*)d_ws;
    size_t off = 0;
    auto nf = [&](size_t e) { float*  p = (float*) (base + off); off += ((e * 4 + 255) / 256) * 256; return p; };
    auto nb = [&](size_t e) { bf16_t* p = (bf16_t*)(base + off); off += ((e * 2 + 255) / 256) * 256; return p; };

    float*  q0     = nf((size_t)B_ * L_ * DIM_);
    float*  y0m    = nf((size_t)B_ * DIM_);
    float*  p5     = nf((size_t)B_ * CIN_ * 25);
    float*  p9     = nf((size_t)B_ * CIN_ * 81);
    float*  p13    = nf((size_t)B_ * CIN_ * 169);
    float*  q5     = nf((size_t)B_ * 25 * DIM_);
    float*  q9     = nf((size_t)B_ * 81 * DIM_);
    float*  q13    = nf((size_t)B_ * 169 * DIM_);
    bf16_t* featb  = nb((size_t)B_ * L_ * DM_);
    bf16_t* wInB   = nb((size_t)E2_ * DM_);
    bf16_t* wXpB   = nb((size_t)K4_ * 96 * DI_);
    bf16_t* wDtB   = nb((size_t)K4_ * DI_ * R_);
    bf16_t* wOutB  = nb((size_t)DM_ * DI_);
    bf16_t* wCbrB  = nb((size_t)DIM_ * DM_);
    float*  xz     = nf((size_t)B_ * L_ * E2_);
    float*  xc     = nf((size_t)B_ * L_ * DI_);
    bf16_t* xs4    = nb((size_t)K4_ * B_ * L_ * DI_);
    float*  xdbl   = nf((size_t)K4_ * B_ * L_ * 96);
    bf16_t* xdblb  = nb((size_t)K4_ * B_ * L_ * 96);
    float*  delta  = nf((size_t)K4_ * B_ * L_ * DI_);
    float*  outy   = nf((size_t)K4_ * B_ * DI_ * L_);
    bf16_t* yln    = nb((size_t)B_ * L_ * DI_);
    float*  yproj  = nf((size_t)B_ * L_ * DM_);
    bf16_t* yprojb = nb((size_t)B_ * L_ * DM_);
    float*  cbrpre = nf((size_t)B_ * L_ * DIM_);
    (void)ws_size; (void)n_in; (void)in_sizes; (void)out_size;

    dim3 gblk(32, 8, 1);        // 8 wave32s per block
    auto cvt = [&](const float* s, bf16_t* d, int n) {
        pm_cvt_bf16<<<(n + 255) / 256, 256, 0, stream>>>(s, d, n);
    };

    // weight conversions
    cvt(in_proj,  wInB,  E2_ * DM_);
    cvt(x_proj,   wXpB,  K4_ * 96 * DI_);
    cvt(dt_w,     wDtB,  K4_ * DI_ * R_);
    cvt(out_proj, wOutB, DM_ * DI_);
    cvt(cbr_w,    wCbrB, DIM_ * DM_);

    // pool pyramid
    pm_pool_conv<<<B_ * L_, DIM_, 0, stream>>>(x, pool_w, pool_b, bn_g, bn_b, bn_m, bn_v, q0, L_, 0);
    pm_pool0_mean<<<B_, DIM_, 0, stream>>>(q0, y0m);
    pm_adap_pool<<<B_ * CIN_, 25, 0, stream>>>(x, p5, 5);
    pm_adap_pool<<<B_ * CIN_, 81, 0, stream>>>(x, p9, 9);
    pm_adap_pool<<<B_ * CIN_, 169, 0, stream>>>(x, p13, 13);
    pm_pool_conv<<<B_ * 25,  DIM_, 0, stream>>>(p5,  pool_w, pool_b, bn_g, bn_b, bn_m, bn_v, q5,  25,  1);
    pm_pool_conv<<<B_ * 81,  DIM_, 0, stream>>>(p9,  pool_w, pool_b, bn_g, bn_b, bn_m, bn_v, q9,  81,  2);
    pm_pool_conv<<<B_ * 169, DIM_, 0, stream>>>(p13, pool_w, pool_b, bn_g, bn_b, bn_m, bn_v, q13, 169, 3);
    pm_feat_assemble<<<B_ * L_, DM_, 0, stream>>>(x, y0m, q5, q9, q13, featb);

    // in_proj GEMM (LDS/async): M=1024, N=4096, K=1024
    pm_gemm_bf16_lds<<<(1024 / 128) * (E2_ / 64), gblk, 0, stream>>>(
        featb, wInB, xz, B_ * L_, E2_, DM_, DM_, DM_);

    pm_dwconv_silu<<<(B_ * L_ * DI_ + 255) / 256, 256, 0, stream>>>(xz, conv_w, conv_b, xc);
    pm_build_xs4<<<(K4_ * B_ * L_ * DI_ + 255) / 256, 256, 0, stream>>>(xc, xs4);

    // x_proj GEMMs (N=96 -> direct kernel): per k, M=1024, N=96, K=2048
    for (int k = 0; k < K4_; k++) {
        int tiles = (B_ * L_ / 16) * (96 / 16);
        pm_gemm_bf16<<<(tiles + 7) / 8, gblk, 0, stream>>>(
            xs4 + (size_t)k * B_ * L_ * DI_, wXpB + (size_t)k * 96 * DI_,
            xdbl + (size_t)k * B_ * L_ * 96, B_ * L_, 96, DI_, DI_, DI_);
    }
    cvt(xdbl, xdblb, K4_ * B_ * L_ * 96);

    // dt GEMMs (LDS/async): per k, M=1024, N=2048, K=64 (lda=96)
    for (int k = 0; k < K4_; k++) {
        pm_gemm_bf16_lds<<<(1024 / 128) * (DI_ / 64), gblk, 0, stream>>>(
            xdblb + (size_t)k * B_ * L_ * 96, wDtB + (size_t)k * DI_ * R_,
            delta + (size_t)k * B_ * L_ * DI_, B_ * L_, DI_, R_, 96, R_);
    }
    pm_softplus_bias<<<(K4_ * B_ * L_ * DI_ + 255) / 256, 256, 0, stream>>>(delta, dt_b);

    // selective scan
    pm_scan<<<dim3(DI_ / 64, K4_ * B_), 64, 0, stream>>>(xs4, delta, xdbl, A_logs, outy);

    pm_combine_ln_gate<<<B_ * L_, 256, 0, stream>>>(outy, xc, xz, Ds, ln_g, ln_b, yln);

    // out_proj GEMM (LDS/async): M=1024, N=1024, K=2048
    pm_gemm_bf16_lds<<<(1024 / 128) * (DM_ / 64), gblk, 0, stream>>>(
        yln, wOutB, yproj, B_ * L_, DM_, DI_, DI_, DI_);
    cvt(yproj, yprojb, B_ * L_ * DM_);

    // cbr GEMM (LDS/async): M=1024, N=128, K=1024
    pm_gemm_bf16_lds<<<(1024 / 128) * (DIM_ / 64), gblk, 0, stream>>>(
        yprojb, wCbrB, cbrpre, B_ * L_, DIM_, DM_, DM_, DM_);

    pm_cbr_finish<<<(B_ * DIM_ * L_ + 255) / 256, 256, 0, stream>>>(
        cbrpre, cbr_b, cbr_g, cbr_bt, cbr_m, cbr_v, (float*)d_out);
}
